// CustomFlashAttention_28381143892581
// MI455X (gfx1250) — compile-verified, matrix operand order checked by reading, not run on hardware
//
#include <hip/hip_runtime.h>
#include <hip/hip_bf16.h>

// ---------------------------------------------------------------------------
// MHA block on MI455X (gfx1250, wave32, WMMA + TDM).
//   x:[2,2048,1024] f32; w_q/w_k/w_v/w_o:[1024,1024] f32; out:[2,2048,1024] f32
// Pipeline: f32->bf16 convert -> QKV WMMA GEMMs (weights TDM-staged in LDS,
//           V stored transposed) -> flash attention (WMMA QK^T + online
//           softmax + WMMA PV) -> output-projection WMMA GEMM (f32 store).
// ---------------------------------------------------------------------------

#define BATCH    2
#define SEQ      2048
#define HIDDEN   1024
#define HEADS    16
#define HEAD_DIM 64

typedef __attribute__((ext_vector_type(16))) __bf16 v16bf;
typedef __attribute__((ext_vector_type(8)))  __bf16 v8bf;
typedef __attribute__((ext_vector_type(8)))  float  v8f;

#if defined(__has_builtin)
#  if __has_builtin(__builtin_amdgcn_tensor_load_to_lds)
#    define HAVE_TDM 1
#  else
#    define HAVE_TDM 0
#  endif
#  if __has_builtin(__builtin_amdgcn_s_wait_tensorcnt)
#    define HAVE_WAIT_TENSOR 1
#  else
#    define HAVE_WAIT_TENSOR 0
#  endif
#else
#  define HAVE_TDM 0
#  define HAVE_WAIT_TENSOR 0
#endif

static __device__ __forceinline__ __bf16 to_bf16(float f) {
  union { float f; unsigned u; } c; c.f = f;
  unsigned r = c.u + 0x7FFFu + ((c.u >> 16) & 1u);   // round-to-nearest-even
  unsigned short h = (unsigned short)(r >> 16);
  return __builtin_bit_cast(__bf16, h);
}

__global__ void convert_f32_bf16(const float* __restrict__ in,
                                 __bf16* __restrict__ out, int n) {
  int i = blockIdx.x * blockDim.x + threadIdx.x;
  int stride = gridDim.x * blockDim.x;
  for (; i < n; i += stride) out[i] = to_bf16(in[i]);
}

// Load one wave32 WMMA 16x16x32 bf16 fragment slice (A layout; B is symmetric
// because all our operands are K-contiguous).
// lane 0-15 : k = k0 + {0..7, 16..23};  lane 16-31 : k = k0 + {8..15, 24..31}
static __device__ __forceinline__ v16bf load_frag(const __bf16* rowptr, int k0, int half) {
  const __bf16* p = rowptr + k0 + half * 8;
  v8bf lo = *(const v8bf*)(p);
  v8bf hi = *(const v8bf*)(p + 16);
  v16bf r;
#pragma unroll
  for (int i = 0; i < 8; ++i) { r[i] = lo[i]; r[8 + i] = hi[i]; }
  return r;
}

static __device__ __forceinline__ v8f wmma_bf16(v16bf a, v16bf b, v8f c) {
  return __builtin_amdgcn_wmma_f32_16x16x32_bf16(false, a, false, b,
                                                 (short)0, c, false, false);
}

// ---------------------------------------------------------------------------
// Tensor Data Mover: load a 64-row x 64-element bf16 tile (row stride 1024
// elements) from global into LDS at lds_byte_off, row-major compacted.
// D# layout per CDNA5 ISA.
// ---------------------------------------------------------------------------
#if HAVE_TDM
typedef __attribute__((ext_vector_type(4))) unsigned int u32x4;
typedef __attribute__((ext_vector_type(4))) int          i32x4;
typedef __attribute__((ext_vector_type(8))) int          i32x8;

static __device__ __forceinline__ void tdm_load_tile_64x64(const __bf16* gsrc,
                                                           unsigned lds_byte_off) {
  unsigned long long ga = (unsigned long long)(size_t)gsrc;
  u32x4 g0;
  g0[0] = 1u;                                   // count=1, no gather
  g0[1] = lds_byte_off;                         // lds_addr [63:32]
  g0[2] = (unsigned)(ga & 0xFFFFFFFFull);       // global_addr lo
  g0[3] = (unsigned)(ga >> 32) | 0x80000000u;   // global_addr hi | type=2<<30
  i32x8 g1;
  g1[0] = 0x00010000;                           // wg_mask=0, data_size=2B
  g1[1] = (int)(1024u << 16);                   // tensor_dim0 = 1024 (lo16)
  g1[2] = (int)(64u << 16);                     // tensor_dim1 = 64, dim0 hi=0
  g1[3] = (int)(64u << 16);                     // tile_dim0 = 64, dim1 hi=0
  g1[4] = 64;                                   // tile_dim1 = 64, tile_dim2=0
  g1[5] = 1024;                                 // tensor_dim0_stride = 1024
  g1[6] = 0;
  g1[7] = 0;
  i32x4 z4 = {};
#if __clang_major__ >= 23
  i32x8 z8 = {};
  __builtin_amdgcn_tensor_load_to_lds(g0, g1, z4, z4, z8, 0);
#else
  __builtin_amdgcn_tensor_load_to_lds(g0, g1, z4, z4, 0);
#endif
}
#endif

static __device__ __forceinline__ void wait_tensorcnt_le(int n) {
#if HAVE_WAIT_TENSOR
  if (n) __builtin_amdgcn_s_wait_tensorcnt((short)1);
  else   __builtin_amdgcn_s_wait_tensorcnt((short)0);
#else
  if (n) asm volatile("s_wait_tensorcnt 0x1" ::: "memory");
  else   asm volatile("s_wait_tensorcnt 0x0" ::: "memory");
#endif
}

// ---------------------------------------------------------------------------
// C[M,1024] = A[M,1024] * W[1024,1024]^T   (bf16 in, f32 accumulate)
// Block = 8 waves = 128 rows x 64 cols. The 64-row weight strip is staged in
// LDS in 64x64-element K-slabs, double-buffered via the TDM (TENSORcnt),
// shared by all 8 waves (8x global-traffic reduction, ds_load_b128 feeds).
// mode 0: store bf16 row-major
// mode 1: store bf16 transposed per head: Vt[b][h][d][s]
// mode 2: store f32 row-major
// ---------------------------------------------------------------------------
__global__ void gemm_xwT_wmma(const __bf16* __restrict__ A,
                              const __bf16* __restrict__ W,
                              void* __restrict__ outp, int M, int mode) {
  __shared__ __align__(16) __bf16 wbuf[2][64 * 64];   // 2 x 8KB W slabs
  const int tid   = threadIdx.x;
  const int lane  = tid & 31;
  const int wave  = tid >> 5;
  const int mgrp  = blockIdx.x >> 4;                  // 128-row group
  const int strip = blockIdx.x & 15;                  // 64-col strip
  const int tile_m = mgrp * 8 + wave;
  const int row_l = lane & 15;
  const int half  = lane >> 4;

  const __bf16* arow = A + (size_t)(tile_m * 16 + row_l) * HIDDEN;
  const __bf16* wstrip = W + (size_t)(strip * 64) * HIDDEN;

  v8f acc0 = {}, acc1 = {}, acc2 = {}, acc3 = {};

  auto issue_slab = [&](int s, int buf) {
#if HAVE_TDM
    if (wave == 0)
      tdm_load_tile_64x64(wstrip + s * 64, (unsigned)(size_t)&wbuf[buf][0]);
#else
    // fallback: cooperative global->LDS copy (512 x 16B chunks, 256 threads)
    for (int c = tid; c < 512; c += 256) {
      int r = c >> 3, seg = c & 7;
      *(v8bf*)&wbuf[buf][r * 64 + seg * 8] =
          *(const v8bf*)(wstrip + (size_t)r * HIDDEN + s * 64 + seg * 8);
    }
#endif
  };

  const int NSLAB = HIDDEN / 64;                      // 16
  issue_slab(0, 0);
  for (int s = 0; s < NSLAB; ++s) {
    const int cur = s & 1;
    if (s + 1 < NSLAB) issue_slab(s + 1, (s + 1) & 1);
#if HAVE_TDM
    if (wave == 0) wait_tensorcnt_le((s + 1 < NSLAB) ? 1 : 0);
#endif
    __syncthreads();                                  // publish wbuf[cur]
    const __bf16* bbase = &wbuf[cur][0];
#pragma unroll
    for (int kk = 0; kk < 64; kk += 32) {
      v16bf a  = load_frag(arow, s * 64 + kk, half);
      v16bf b0 = load_frag(bbase + ( 0 + row_l) * 64, kk, half);
      v16bf b1 = load_frag(bbase + (16 + row_l) * 64, kk, half);
      v16bf b2 = load_frag(bbase + (32 + row_l) * 64, kk, half);
      v16bf b3 = load_frag(bbase + (48 + row_l) * 64, kk, half);
      acc0 = wmma_bf16(a, b0, acc0);
      acc1 = wmma_bf16(a, b1, acc1);
      acc2 = wmma_bf16(a, b2, acc2);
      acc3 = wmma_bf16(a, b3, acc3);
    }
    __syncthreads();                                  // protect next overwrite
  }

  v8f accs[4] = {acc0, acc1, acc2, acc3};
#pragma unroll
  for (int t = 0; t < 4; ++t) {
#pragma unroll
    for (int r = 0; r < 8; ++r) {
      int m = tile_m * 16 + r + 8 * half;             // C layout: VGPR r -> row
      int n = strip * 64 + t * 16 + row_l;            // lane -> col within tile
      float v = accs[t][r];
      if (mode == 2) {
        ((float*)outp)[(size_t)m * HIDDEN + n] = v;
      } else if (mode == 1) {
        int b = m / SEQ, s2 = m % SEQ;
        int h = n / HEAD_DIM, d = n % HEAD_DIM;
        ((__bf16*)outp)[(((size_t)(b * HEADS + h) * HEAD_DIM) + d) * SEQ + s2] = to_bf16(v);
      } else {
        ((__bf16*)outp)[(size_t)m * HIDDEN + n] = to_bf16(v);
      }
    }
  }
}

// ---------------------------------------------------------------------------
// Flash attention: one wave per 16-query tile per (b,h); key blocks of 32.
// Q,K row-major bf16 [B*S,1024]; V transposed bf16 [B][H][64][S].
// Output attn bf16 row-major [B*S,1024].
// ---------------------------------------------------------------------------
__global__ void flash_attn_wmma(const __bf16* __restrict__ Q,
                                const __bf16* __restrict__ Kmat,
                                const __bf16* __restrict__ Vt,
                                __bf16* __restrict__ O) {
  __shared__ __align__(16) __bf16 pshare[4][16 * 32];  // P tile per wave
  const int lane  = threadIdx.x & 31;
  const int wave  = threadIdx.x >> 5;
  const int bh    = blockIdx.x >> 5;                  // 32 blocks per (b,h)
  const int qt    = (blockIdx.x & 31) * 4 + wave;     // q tile index
  const int b     = bh >> 4;
  const int h     = bh & 15;
  const int row_l = lane & 15;
  const int half  = lane >> 4;
  const int q0    = qt * 16;

  const __bf16* qrow = Q + (size_t)(b * SEQ + q0 + row_l) * HIDDEN + h * HEAD_DIM;
  v16bf qa0 = load_frag(qrow, 0, half);   // k-dim 0..31  of head
  v16bf qa1 = load_frag(qrow, 32, half);  // k-dim 32..63 of head

  float m_run[8], l_run[8];
#pragma unroll
  for (int r = 0; r < 8; ++r) { m_run[r] = -1e30f; l_run[r] = 0.f; }
  v8f o0 = {}, o1 = {}, o2 = {}, o3 = {};

  const float scale = 0.125f;  // 1/sqrt(64)
  __bf16* prow = &pshare[wave][row_l * 32];
  const __bf16* vbase = Vt + (size_t)((b * HEADS + h) * HEAD_DIM) * SEQ;

  for (int j = 0; j < SEQ / 32; ++j) {
    // ---- scores: 16x32 = Q(16x64) . K_block(32x64)^T  (4 WMMAs) ----
    const __bf16* k0row = Kmat + (size_t)(b * SEQ + j * 32 +  0 + row_l) * HIDDEN + h * HEAD_DIM;
    const __bf16* k1row = Kmat + (size_t)(b * SEQ + j * 32 + 16 + row_l) * HIDDEN + h * HEAD_DIM;
    v8f s0 = {}, s1 = {};
    s0 = wmma_bf16(qa0, load_frag(k0row, 0, half), s0);
    s0 = wmma_bf16(qa1, load_frag(k0row, 32, half), s0);
    s1 = wmma_bf16(qa0, load_frag(k1row, 0, half), s1);
    s1 = wmma_bf16(qa1, load_frag(k1row, 32, half), s1);

    // ---- online softmax; P written to LDS (transpose C-layout -> A-layout) ----
#pragma unroll
    for (int r = 0; r < 8; ++r) {
      float v0 = s0[r] * scale, v1 = s1[r] * scale;
      float mx = fmaxf(v0, v1);
#pragma unroll
      for (int off = 8; off >= 1; off >>= 1)
        mx = fmaxf(mx, __shfl_xor(mx, off, 32));      // reduce across 16-lane half
      float mnew = fmaxf(m_run[r], mx);
      float p0 = __expf(v0 - mnew);
      float p1 = __expf(v1 - mnew);
      float rs = p0 + p1;
#pragma unroll
      for (int off = 8; off >= 1; off >>= 1)
        rs += __shfl_xor(rs, off, 32);
      float alpha = __expf(m_run[r] - mnew);
      m_run[r] = mnew;
      l_run[r] = l_run[r] * alpha + rs;
      o0[r] *= alpha; o1[r] *= alpha; o2[r] *= alpha; o3[r] *= alpha;
      pshare[wave][(r + 8 * half) * 32 + row_l]      = to_bf16(p0);
      pshare[wave][(r + 8 * half) * 32 + 16 + row_l] = to_bf16(p1);
    }

    // ---- PV: O(16x64) += P(16x32) . V_block(32x64)  (4 WMMAs) ----
    // per-wave LDS region; DS ops of one wave are in-order (RAW safe).
    v16bf pa  = load_frag(prow, 0, half);
    v16bf vb0 = load_frag(vbase + (size_t)( 0 + row_l) * SEQ + j * 32, 0, half);
    v16bf vb1 = load_frag(vbase + (size_t)(16 + row_l) * SEQ + j * 32, 0, half);
    v16bf vb2 = load_frag(vbase + (size_t)(32 + row_l) * SEQ + j * 32, 0, half);
    v16bf vb3 = load_frag(vbase + (size_t)(48 + row_l) * SEQ + j * 32, 0, half);
    o0 = wmma_bf16(pa, vb0, o0);
    o1 = wmma_bf16(pa, vb1, o1);
    o2 = wmma_bf16(pa, vb2, o2);
    o3 = wmma_bf16(pa, vb3, o3);
  }

  // ---- normalize and store ----
  v8f os[4] = {o0, o1, o2, o3};
#pragma unroll
  for (int t = 0; t < 4; ++t) {
#pragma unroll
    for (int r = 0; r < 8; ++r) {
      int m = q0 + r + 8 * half;
      int n = h * HEAD_DIM + t * 16 + row_l;
      O[(size_t)(b * SEQ + m) * HIDDEN + n] = to_bf16(os[t][r] / l_run[r]);
    }
  }
}

// ---------------------------------------------------------------------------
extern "C" void kernel_launch(void* const* d_in, const int* in_sizes, int n_in,
                              void* d_out, int out_size, void* d_ws, size_t ws_size,
                              hipStream_t stream) {
  const float* x  = (const float*)d_in[0];
  const float* wq = (const float*)d_in[1];
  const float* wk = (const float*)d_in[2];
  const float* wv = (const float*)d_in[3];
  const float* wo = (const float*)d_in[4];

  const size_t NX = (size_t)BATCH * SEQ * HIDDEN;   // 4,194,304
  const size_t NW = (size_t)HIDDEN * HIDDEN;        // 1,048,576

  char* ws = (char*)d_ws;
  size_t off = 0;
  __bf16* xb  = (__bf16*)(ws + off); off += NX * 2;
  __bf16* wqb = (__bf16*)(ws + off); off += NW * 2;
  __bf16* wkb = (__bf16*)(ws + off); off += NW * 2;
  __bf16* wvb = (__bf16*)(ws + off); off += NW * 2;
  __bf16* wob = (__bf16*)(ws + off); off += NW * 2;
  __bf16* qb  = (__bf16*)(ws + off); off += NX * 2;
  __bf16* kb  = (__bf16*)(ws + off); off += NX * 2;
  __bf16* vtb = (__bf16*)(ws + off); off += NX * 2;   // [B][H][64][S]
  __bf16* atb = (__bf16*)(ws + off); off += NX * 2;

  convert_f32_bf16<<<2048, 256, 0, stream>>>(x,  xb,  (int)NX);
  convert_f32_bf16<<<1024, 256, 0, stream>>>(wq, wqb, (int)NW);
  convert_f32_bf16<<<1024, 256, 0, stream>>>(wk, wkb, (int)NW);
  convert_f32_bf16<<<1024, 256, 0, stream>>>(wv, wvb, (int)NW);
  convert_f32_bf16<<<1024, 256, 0, stream>>>(wo, wob, (int)NW);

  const int M = BATCH * SEQ;                        // 4096 rows
  const int gemm_blocks = (M / 128) * 16;           // 512 (128-row x 64-col tiles)
  gemm_xwT_wmma<<<gemm_blocks, 256, 0, stream>>>(xb, wqb, qb,  M, 0);
  gemm_xwT_wmma<<<gemm_blocks, 256, 0, stream>>>(xb, wkb, kb,  M, 0);
  gemm_xwT_wmma<<<gemm_blocks, 256, 0, stream>>>(xb, wvb, vtb, M, 1);

  // BATCH*HEADS*(SEQ/16/4) = 1024 blocks, 4 waves each
  flash_attn_wmma<<<BATCH * HEADS * (SEQ / 16 / 4), 128, 0, stream>>>(qb, kb, vtb, atb);

  gemm_xwT_wmma<<<gemm_blocks, 256, 0, stream>>>(atb, wob, d_out, M, 2);
}
